// TransformerModel_40209483825350
// MI455X (gfx1250) — matrix operand order repaired and checked by reading, not compiled, not run on hardware
//
#include <hip/hip_runtime.h>

// ---------------------------------------------------------------------------
// Problem dimensions (fixed by the reference)
// ---------------------------------------------------------------------------
#define NN   12
#define BB   4096
#define EE   256
#define HH   8
#define DHD  64          // 2E/H
#define NHID 1024
#define LL   4
#define GNUM 8
#define NGF  64          // FEAT - 3*GNUM = 88 - 24
#define FEAT 88
#define NB   (NN*BB)         // 49152
#define NB3  (NB*3)          // 147456
#define SQRTE 16.0f

typedef __attribute__((ext_vector_type(16))) _Float16 v16h;
typedef __attribute__((ext_vector_type(8)))  _Float16 v8h;
typedef __attribute__((ext_vector_type(4)))  _Float16 v4h;
typedef __attribute__((ext_vector_type(8)))  float    v8f;

// ---------------------------------------------------------------------------
// Generic WMMA GEMM:  C = epi(A[M,K] @ W[K,Nc])
//   epilogue: v = AB + bias[col]; if ADD v += C_old; if RELU v=max(v,0);
//             v *= alpha; if rowdiv v /= rowdiv[row];
// Block tile 128x64, 8 waves (4x2), each wave a 32x32 tile (2x2 WMMA frags).
// A and W are fp32 in HBM, converted to f16 while staging into LDS.
// Fast path (full tile): float4 global loads + packed f32->f16 converts.
// Edge path: clamped-address loads + select (no exec-mask churn).
// ---------------------------------------------------------------------------
#define BM 128
#define BN 64
#define BK 32
#define KST 40   // BK + 8 halfs padding: 80B rows, 16B aligned sub-chunks

#define FLAG_RELU 1
#define FLAG_ADD  2

__global__ __launch_bounds__(256) void gemm_wmma(
    const float* __restrict__ A, long lda,
    const float* __restrict__ W, int ldw,
    const float* __restrict__ bias,
    float* __restrict__ C, long ldc,
    const float* __restrict__ rowdiv,
    int M, int Nc, int K, float alpha, int flags)
{
  __shared__ __align__(16) _Float16 As[BM * KST];
  __shared__ __align__(16) _Float16 Bs[BN * KST];

  const int t    = threadIdx.x;
  const int lane = t & 31;
  const int wave = t >> 5;
  const int wr   = wave >> 1;     // 0..3 : wave row
  const int wc   = wave & 1;      // 0..1 : wave col
  const int lm   = lane & 15;     // M/N within a 16-wide fragment
  const int kh   = lane >> 4;     // K-half selector
  const long rowBase = (long)blockIdx.y * BM;
  const int  colBase = blockIdx.x * BN;

  const bool fullA = (rowBase + BM <= (long)M);
  const bool fullB = (colBase + BN <= Nc);

  v8f c00, c01, c10, c11;
#pragma unroll
  for (int i = 0; i < 8; ++i) { c00[i]=0.f; c01[i]=0.f; c10[i]=0.f; c11[i]=0.f; }

  for (int kb = 0; kb < K; kb += BK) {
    const bool fullK = (kb + BK <= K);

    // ---- stage A tile (BM x BK), k contiguous per row ----
    if (fullA && fullK) {
      // 1024 float4 chunks; 4 per thread; unguarded 128-bit loads
#pragma unroll
      for (int j = 0; j < 4; ++j) {
        int idx = j * 256 + t;
        int m  = idx >> 3;            // row 0..127
        int kq = (idx & 7) << 2;      // k quad 0,4,..,28
        const float4 vv = *(const float4*)(A + (rowBase + m) * lda + kb + kq);
        auto lo = __builtin_amdgcn_cvt_pkrtz(vv.x, vv.y);  // __fp16 x2
        auto hi = __builtin_amdgcn_cvt_pkrtz(vv.z, vv.w);
        v4h o;
        o[0] = (_Float16)lo[0]; o[1] = (_Float16)lo[1];
        o[2] = (_Float16)hi[0]; o[3] = (_Float16)hi[1];
        *(v4h*)(As + m * KST + kq) = o;
      }
    } else {
#pragma unroll
      for (int j = 0; j < 16; ++j) {
        int idx = j * 256 + t;
        int m = idx >> 5, k = idx & 31;
        long r  = rowBase + m;
        long rc = r < (long)M ? r : (long)(M - 1);
        int  kc = (kb + k) < K ? (kb + k) : (K - 1);
        float v = A[rc * lda + kc];                  // always in-range address
        if (r >= (long)M || (kb + k) >= K) v = 0.f;  // select, not branch
        As[m * KST + k] = (_Float16)v;
      }
    }

    // ---- stage B tile (BK x BN) transposed into Bs[n][k] ----
    if (fullB && fullK) {
      // 512 float4 chunks; 2 per thread; unguarded 128-bit loads
#pragma unroll
      for (int j = 0; j < 2; ++j) {
        int idx = j * 256 + t;
        int k  = idx >> 4;            // 0..31
        int nq = (idx & 15) << 2;     // n quad 0,4,..,60
        const float4 vv = *(const float4*)(W + (long)(kb + k) * ldw + colBase + nq);
        Bs[(nq + 0) * KST + k] = (_Float16)vv.x;
        Bs[(nq + 1) * KST + k] = (_Float16)vv.y;
        Bs[(nq + 2) * KST + k] = (_Float16)vv.z;
        Bs[(nq + 3) * KST + k] = (_Float16)vv.w;
      }
    } else {
#pragma unroll
      for (int j = 0; j < 8; ++j) {
        int idx = j * 256 + t;
        int n = idx & 63, k = idx >> 6;
        int cc = colBase + n;
        int cl = cc < Nc ? cc : (Nc - 1);
        int kc = (kb + k) < K ? (kb + k) : (K - 1);
        float v = W[(long)kc * ldw + cl];
        if (cc >= Nc || (kb + k) >= K) v = 0.f;
        Bs[n * KST + k] = (_Float16)v;
      }
    }
    if (kb + BK < K)
      __builtin_prefetch(A + rowBase * lda + kb + BK, 0, 1);
    __syncthreads();

    // ---- load fragments ----
    // A (16x32 f16): lane(kh,lm) -> row lm, K = {kh*8..+7} U {16+kh*8..+7}
    const _Float16* a0p = &As[(wr * 32 +      lm) * KST];
    const _Float16* a1p = &As[(wr * 32 + 16 + lm) * KST];
    v8h a0lo = *(const v8h*)(a0p + kh * 8);
    v8h a0hi = *(const v8h*)(a0p + 16 + kh * 8);
    v8h a1lo = *(const v8h*)(a1p + kh * 8);
    v8h a1hi = *(const v8h*)(a1p + 16 + kh * 8);
    // B (32x16 f16): lane(kh,lm) -> col lm, K = {kh*8..+7} U {16+kh*8..+7}
    const _Float16* b0p = &Bs[(wc * 32 +      lm) * KST];
    const _Float16* b1p = &Bs[(wc * 32 + 16 + lm) * KST];
    v8h b0lo = *(const v8h*)(b0p + kh * 8);
    v8h b0hi = *(const v8h*)(b0p + 16 + kh * 8);
    v8h b1lo = *(const v8h*)(b1p + kh * 8);
    v8h b1hi = *(const v8h*)(b1p + 16 + kh * 8);

    v16h a0, a1, b0, b1;
#pragma unroll
    for (int i = 0; i < 8; ++i) {
      a0[i] = a0lo[i]; a0[i + 8] = a0hi[i];
      a1[i] = a1lo[i]; a1[i + 8] = a1hi[i];
      b0[i] = b0lo[i]; b0[i + 8] = b0hi[i];
      b1[i] = b1lo[i]; b1[i + 8] = b1hi[i];
    }

    c00 = __builtin_amdgcn_wmma_f32_16x16x32_f16(false, a0, false, b0, (short)0, c00, false, false);
    c01 = __builtin_amdgcn_wmma_f32_16x16x32_f16(false, a0, false, b1, (short)0, c01, false, false);
    c10 = __builtin_amdgcn_wmma_f32_16x16x32_f16(false, a1, false, b0, (short)0, c10, false, false);
    c11 = __builtin_amdgcn_wmma_f32_16x16x32_f16(false, a1, false, b1, (short)0, c11, false, false);
    __syncthreads();
  }

  // ---- epilogue + store ----
  v8f cts[4] = {c00, c01, c10, c11};
#pragma unroll
  for (int tt = 0; tt < 4; ++tt) {
    int ti = tt >> 1, tj = tt & 1;
    long r0  = rowBase + wr * 32 + ti * 16;
    int  col = colBase + wc * 32 + tj * 16 + lm;
    if (col >= Nc) continue;
    float bval = bias ? bias[col] : 0.f;
#pragma unroll
    for (int j = 0; j < 8; ++j) {
      long r = r0 + j + 8 * kh;
      if (r >= M) continue;
      float v = cts[tt][j] + bval;
      if (flags & FLAG_ADD)  v += C[r * ldc + col];
      if (flags & FLAG_RELU) v = v > 0.f ? v : 0.f;
      v *= alpha;
      if (rowdiv) v /= rowdiv[r];
      C[r * ldc + col] = v;
    }
  }
}

// ---------------------------------------------------------------------------
// Small support kernels
// ---------------------------------------------------------------------------

// Build g0 (NB,3,GNUM) [transposed from src] and gdir (NB,3,2)
__global__ void prep_kernel(const float* __restrict__ src,
                            float* __restrict__ g0, float* __restrict__ gdir)
{
  long idx = (long)blockIdx.x * 256 + threadIdx.x; // over NB*3
  if (idx >= (long)NB3) return;
  long item = idx / 3;
  int  x    = (int)(idx % 3);
  const float* s = src + item * FEAT;
  float* g = g0 + idx * GNUM;
#pragma unroll
  for (int gg = 0; gg < GNUM; ++gg) g[gg] = s[gg * 3 + x];
  gdir[idx * 2 + 0] = s[1 * 3 + x];
  gdir[idx * 2 + 1] = s[2 * 3 + x];
}

// ng_src[row, c] += concat(pos0[trav0[n]], pos1[trav1[n]])[c]
__global__ void posadd_kernel(float* __restrict__ ng,
                              const float* __restrict__ pos0,
                              const float* __restrict__ pos1,
                              const int* __restrict__ tr0,
                              const int* __restrict__ tr1)
{
  long idx = (long)blockIdx.x * 256 + threadIdx.x;
  if (idx >= (long)NB * EE) return;
  long row = idx >> 8;
  int  c   = (int)(idx & 255);
  int  n   = (int)(row / BB);
  float p = (c < 128) ? pos0[tr0[n] * 128 + c] : pos1[tr1[n] * 128 + (c - 128)];
  ng[idx] += p;
}

// maskb[h,n,m] = relB[h] + sum_r relation[m,n,r]*relW[r,h]
__global__ void relb_kernel(const float* __restrict__ rel,
                            const float* __restrict__ relW,
                            const float* __restrict__ relB,
                            float* __restrict__ maskb)
{
  int idx = blockIdx.x * 256 + threadIdx.x;
  if (idx >= HH * NN * NN) return;
  int h = idx / (NN * NN), rem = idx % (NN * NN);
  int n = rem / NN, m = rem % NN;
  float s = relB[h];
#pragma unroll
  for (int r = 0; r < 4; ++r) s += rel[(m * NN + n) * 4 + r] * relW[r * HH + h];
  maskb[idx] = s;
}

// dst32[item3*32 + 30 + j] = gdir[item3*2 + j]
__global__ void gdirfill_kernel(float* __restrict__ dst, const float* __restrict__ gdir)
{
  long idx = (long)blockIdx.x * 256 + threadIdx.x;
  if (idx >= (long)NB3 * 2) return;
  long item3 = idx >> 1;
  int  j = (int)(idx & 1);
  dst[item3 * 32 + 30 + j] = gdir[item3 * 2 + j];
}

// gram[item, a, b] = sum_x gp[item, x, a]*gp[item, x, b];  fn = sqrt(sum gram^2)+1
__global__ __launch_bounds__(256) void gram_kernel(const float* __restrict__ gp,
                                                   float* __restrict__ gram,
                                                   float* __restrict__ fn)
{
  __shared__ float P[96];
  __shared__ float red[256];
  long item = blockIdx.x;
  int  t    = threadIdx.x;
  if (t < 96) P[t] = gp[item * 96 + t];
  __syncthreads();
  float ss = 0.f;
#pragma unroll
  for (int e = t; e < 1024; e += 256) {
    int a = e >> 5, b2 = e & 31;
    float v = P[a] * P[b2] + P[32 + a] * P[32 + b2] + P[64 + a] * P[64 + b2];
    gram[item * 1024 + e] = v;
    ss += v * v;
  }
  red[t] = ss;
  __syncthreads();
  for (int s = 128; s > 0; s >>= 1) { if (t < s) red[t] += red[t + s]; __syncthreads(); }
  if (t == 0) fn[item] = sqrtf(red[0]) + 1.f;
}

// strided copy: dst[row*ldd + c] = src[row*lds + c]
__global__ void copy_strided(const float* __restrict__ src, long lds_,
                             float* __restrict__ dst, long ldd,
                             long rows, int cols)
{
  long idx = (long)blockIdx.x * 256 + threadIdx.x;
  if (idx >= rows * cols) return;
  long row = idx / cols;
  int  c   = (int)(idx % cols);
  dst[row * ldd + c] = src[row * lds_ + c];
}

__global__ void add_kernel(float* __restrict__ dst, const float* __restrict__ src, long n)
{
  long idx = (long)blockIdx.x * 256 + threadIdx.x;
  if (idx < n) dst[idx] += src[idx];
}

// LayerNorm over E=256 with optional residual:  out = LN(xin (+ addv)) * w + b
__global__ __launch_bounds__(256) void ln_kernel(const float* __restrict__ xin,
                                                 const float* __restrict__ addv,
                                                 const float* __restrict__ w,
                                                 const float* __restrict__ bp,
                                                 float* __restrict__ out)
{
  __shared__ float red[256];
  long row = blockIdx.x;
  int  t   = threadIdx.x;
  float x = xin[row * EE + t];
  if (addv) x += addv[row * EE + t];
  red[t] = x; __syncthreads();
  for (int s = 128; s > 0; s >>= 1) { if (t < s) red[t] += red[t + s]; __syncthreads(); }
  float mu = red[0] * (1.f / EE);
  __syncthreads();
  float d = x - mu;
  red[t] = d * d; __syncthreads();
  for (int s = 128; s > 0; s >>= 1) { if (t < s) red[t] += red[t + s]; __syncthreads(); }
  float var = red[0] * (1.f / EE);
  out[row * EE + t] = d * rsqrtf(var + 1e-5f) * w[t] + bp[t];
}

// Attention: one block per (h,b). scores/softmax over N=12, apply to v and vg.
__global__ __launch_bounds__(192) void attn_kernel(
    const float* __restrict__ qh, const float* __restrict__ kk,
    const float* __restrict__ vh, const float* __restrict__ vg,
    const float* __restrict__ gdir, const float* __restrict__ maskb,
    float* __restrict__ attout, float* __restrict__ gatt, int useMask)
{
  const int h = blockIdx.x, b = blockIdx.y, t = threadIdx.x;
  __shared__ float qs[NN * DHD], ks[NN * DHD], vs[NN * DHD], at[NN * NN];

  for (int idx = t; idx < NN * DHD; idx += 192) {
    int n = idx >> 6, d = idx & 63;
    long base = ((long)(n * BB + b)) * 512 + h * 64 + d;
    qs[idx] = qh[base]; ks[idx] = kk[base]; vs[idx] = vh[base];
  }
  __syncthreads();
  for (int idx = t; idx < NN * NN; idx += 192) {
    int n = idx / NN, m = idx % NN;
    float s = 0.f;
#pragma unroll
    for (int d = 0; d < DHD; ++d) s += qs[n * DHD + d] * ks[m * DHD + d];
    s *= 0.125f; // DH^-0.5
    if (useMask) s += maskb[h * (NN * NN) + n * NN + m];
    at[idx] = s;
  }
  __syncthreads();
  if (t < NN) {
    float mx = -1e30f;
    for (int m = 0; m < NN; ++m) mx = fmaxf(mx, at[t * NN + m]);
    float sum = 0.f;
    for (int m = 0; m < NN; ++m) { float e = __expf(at[t * NN + m] - mx); at[t * NN + m] = e; sum += e; }
    float inv = 1.f / sum;
    for (int m = 0; m < NN; ++m) at[t * NN + m] *= inv;
  }
  __syncthreads();
  for (int idx = t; idx < NN * DHD; idx += 192) {
    int n = idx >> 6, d = idx & 63;
    float o = 0.f;
#pragma unroll
    for (int m = 0; m < NN; ++m) o += at[n * NN + m] * vs[m * DHD + d];
    attout[((long)(n * BB + b)) * 512 + h * 64 + d] = o;
  }
  for (int idx = t; idx < NN * 3 * DHD; idx += 192) {
    int n = idx / 192, rem = idx % 192, x = rem >> 6, d = rem & 63;
    float o = 0.f;
    if (d < DHD - 2) {
#pragma unroll
      for (int m = 0; m < NN; ++m)
        o += at[n * NN + m] * vg[(((long)(m * BB + b)) * 3 + x) * 496 + h * 62 + d];
    } else {
#pragma unroll
      for (int m = 0; m < NN; ++m)
        o += at[n * NN + m] * gdir[(((long)(m * BB + b)) * 3 + x) * 2 + (d - 62)];
    }
    gatt[(((long)(n * BB + b)) * 3 + x) * 512 + h * 64 + d] = o;
  }
}

// g3e[item,x,c] = sum_a g3b[item,x,a] * mat3[item,a,c]
__global__ void mat3_apply(const float* __restrict__ g3b,
                           const float* __restrict__ mat3,
                           float* __restrict__ g3e)
{
  long idx = (long)blockIdx.x * 256 + threadIdx.x;
  if (idx >= (long)NB3 * 32) return;
  long item3 = idx >> 5;
  int  c     = (int)(idx & 31);
  long item  = item3 / 3;
  float v = 0.f;
#pragma unroll
  for (int a = 0; a < 32; ++a)
    v += g3b[item3 * 32 + a] * mat3[item * 1024 + a * 32 + c];
  g3e[item3 * 32 + c] = v;
}

// out[row] = (sum_c cat[row,c]*decW[c] + decB) / fn[row]
__global__ __launch_bounds__(128) void decode_kernel(const float* __restrict__ cat,
                                                     const float* __restrict__ decW,
                                                     const float* __restrict__ decB,
                                                     const float* __restrict__ fn,
                                                     float* __restrict__ out)
{
  __shared__ float red[128];
  long row = blockIdx.x;
  int  t   = threadIdx.x;
  float s = 0.f;
  for (int c = t; c < 512; c += 128) s += cat[row * 512 + c] * decW[c];
  red[t] = s; __syncthreads();
  for (int k = 64; k > 0; k >>= 1) { if (t < k) red[t] += red[t + k]; __syncthreads(); }
  if (t == 0) out[row] = (red[0] + decB[0]) / fn[row];
}

// ---------------------------------------------------------------------------
// Host-side orchestration
// ---------------------------------------------------------------------------
enum {
  IN_SRC = 0, IN_TRAV0, IN_TRAV1, IN_REL,
  P_aqW, P_aqB, P_akW, P_akB, P_avW, P_avB, P_avgW, P_angoW, P_angoB, P_agoW,
  P_agpW, P_ag1W, P_ag1B, P_ag2W, P_ag2B, P_gp2W, P_gp3W, P_lg1W, P_lg1B,
  P_lg2W, P_lg2B, P_l1W, P_l1B, P_l2W, P_l2B, P_l3W, P_l3B, P_l4W, P_l4B,
  P_l5W, P_n1W, P_n1B, P_n2W, P_n2B,
  P_pos0, P_pos1, P_relW, P_relB, P_gencW, P_encW, P_encB,
  P_ggW, P_g1W, P_g1B, P_g2W, P_g2B, P_ng1W, P_ng1B, P_ng2W, P_ng2B,
  P_decW, P_decB, P_fnW, P_fnB
};

static inline int cdiv(long a, int b) { return (int)((a + b - 1) / b); }

extern "C" void kernel_launch(void* const* d_in, const int* in_sizes, int n_in,
                              void* d_out, int out_size, void* d_ws, size_t ws_size,
                              hipStream_t stream)
{
  (void)in_sizes; (void)n_in; (void)out_size; (void)ws_size;

  const float* src = (const float*)d_in[IN_SRC];
  const int*   tr0 = (const int*)d_in[IN_TRAV0];
  const int*   tr1 = (const int*)d_in[IN_TRAV1];
  const float* rel = (const float*)d_in[IN_REL];
  auto P = [&](int i) { return (const float*)d_in[i]; };

  // ----- workspace carve (floats) -----
  float* wsf = (float*)d_ws;
  size_t off = 0;
  auto alloc = [&](size_t n) { float* p = wsf + off; off += n; return p; };
  float* g_src  = alloc((size_t)NB3 * EE);
  float* g1     = alloc((size_t)NB3 * EE);
  float* ng_src = alloc((size_t)NB * EE);
  float* xbuf   = alloc((size_t)NB * 512);
  float* qh     = alloc((size_t)NB * 512);
  float* kh     = alloc((size_t)NB * 512);
  float* vh     = alloc((size_t)NB * 512);
  float* attout = alloc((size_t)NB * 512);   // also final concat buffer
  float* gatt   = alloc((size_t)NB3 * 512);
  float* vg     = alloc((size_t)NB3 * 496);
  float* gp     = alloc((size_t)NB3 * 32);
  float* g3b    = alloc((size_t)NB3 * 32);
  float* gram   = alloc((size_t)NB * 1024);  // also mat3
  float* hid    = alloc((size_t)NB * 1024);
  float* fn     = alloc((size_t)NB);
  float* tA     = alloc((size_t)NB * EE);
  float* g0     = alloc((size_t)NB * 3 * GNUM);
  float* gdir   = alloc((size_t)NB * 3 * 2);
  float* maskb  = alloc((size_t)HH * NN * NN);

  auto gemm = [&](const float* A, long lda, const float* W, int ldw,
                  const float* bias, float* C, long ldc, const float* rowdiv,
                  int M, int Nc, int K, float alpha, int flags) {
    dim3 g(cdiv(Nc, BN), cdiv(M, BM));
    gemm_wmma<<<g, 256, 0, stream>>>(A, lda, W, ldw, bias, C, ldc, rowdiv,
                                     M, Nc, K, alpha, flags);
  };

  // ----- preamble -----
  prep_kernel<<<cdiv(NB3, 256), 256, 0, stream>>>(src, g0, gdir);
  // g_src = g0 @ gencW * sqrt(E)
  gemm(g0, GNUM, P(P_gencW), EE, nullptr, g_src, EE, nullptr, NB3, EE, GNUM, SQRTE, 0);
  // ng_src = (ng0 @ encW + encB) * sqrt(E)   (ng0 read strided from src)
  gemm(src + GNUM * 3, FEAT, P(P_encW), EE, P(P_encB), ng_src, EE, nullptr, NB, EE, NGF, SQRTE, 0);
  posadd_kernel<<<cdiv((long)NB * EE, 256), 256, 0, stream>>>(ng_src, P(P_pos0), P(P_pos1), tr0, tr1);
  relb_kernel<<<cdiv(HH * NN * NN, 256), 256, 0, stream>>>(rel, P(P_relW), P(P_relB), maskb);

  // per-layer parameter pointers
  auto PL = [&](int idx, size_t cnt, int li) { return P(idx) + (size_t)li * cnt; };

  for (int li = 0; li < LL; ++li) {
    const float* aqW  = PL(P_aqW, 512u*512, li); const float* aqB  = PL(P_aqB, 512, li);
    const float* akW  = PL(P_akW, 512u*512, li); const float* akB  = PL(P_akB, 512, li);
    const float* avW  = PL(P_avW, 512u*512, li); const float* avB  = PL(P_avB, 512, li);
    const float* avgW = PL(P_avgW, 256u*496, li);
    const float* angoW= PL(P_angoW,512u*256, li); const float* angoB= PL(P_angoB,256, li);
    const float* agoW = PL(P_agoW, 512u*256, li);
    const float* agpW = PL(P_agpW, 256u*30, li);
    const float* ag1W = PL(P_ag1W, 1024u*512, li); const float* ag1B = PL(P_ag1B, 512, li);
    const float* ag2W = PL(P_ag2W, 512u*256, li);  const float* ag2B = PL(P_ag2B, 256, li);
    const float* gp2W = PL(P_gp2W, 256u*30, li);
    const float* gp3W = PL(P_gp3W, 256u*30, li);
    const float* lg1W = PL(P_lg1W, 1024u*1024, li); const float* lg1B = PL(P_lg1B, 1024, li);
    const float* lg2W = PL(P_lg2W, 1024u*256, li);  const float* lg2B = PL(P_lg2B, 256, li);
    const float* l1W  = PL(P_l1W, 512u*1024, li);   const float* l1B  = PL(P_l1B, 1024, li);
    const float* l2W  = PL(P_l2W, 1024u*256, li);   const float* l2B  = PL(P_l2B, 256, li);
    const float* l3W  = PL(P_l3W, 512u*1024, li);   const float* l3B  = PL(P_l3B, 1024, li);
    const float* l4W  = PL(P_l4W, 1024u*1024, li);  const float* l4B  = PL(P_l4B, 1024, li);
    const float* l5W  = PL(P_l5W, 32u*256, li);
    const float* n1W  = PL(P_n1W, 256, li); const float* n1B = PL(P_n1B, 256, li);
    const float* n2W  = PL(P_n2W, 256, li); const float* n2B = PL(P_n2B, 256, li);

    // --- gram feature on g_src -> inv, then x = [inv | ng_src] ---
    gemm(g_src, EE, agpW, 30, nullptr, gp, 32, nullptr, NB3, 30, EE, 1.f, 0);
    gdirfill_kernel<<<cdiv((long)NB3 * 2, 256), 256, 0, stream>>>(gp, gdir);
    gram_kernel<<<NB, 256, 0, stream>>>(gp, gram, fn);
    gemm(gram, 1024, ag1W, 512, ag1B, hid, 512, nullptr, NB, 512, 1024, 1.f, FLAG_RELU);
    gemm(hid, 512, ag2W, 256, ag2B, xbuf, 512, fn, NB, 256, 512, 1.f, 0);
    copy_strided<<<cdiv((long)NB * EE, 256), 256, 0, stream>>>(ng_src, EE, xbuf + EE, 512, NB, EE);

    // --- q,k,v  and geometric v ---
    gemm(xbuf, 512, aqW, 512, aqB, qh, 512, nullptr, NB, 512, 512, 1.f, 0);
    gemm(xbuf, 512, akW, 512, akB, kh, 512, nullptr, NB, 512, 512, 1.f, 0);
    gemm(xbuf, 512, avW, 512, avB, vh, 512, nullptr, NB, 512, 512, 1.f, 0);
    gemm(g_src, EE, avgW, 496, nullptr, vg, 496, nullptr, NB3, 496, EE, 1.f, 0);

    // --- attention ---
    attn_kernel<<<dim3(HH, BB), 192, 0, stream>>>(qh, kh, vh, vg, gdir, maskb,
                                                  attout, gatt, li == 0 ? 1 : 0);

    // --- projections + residuals ---
    gemm(attout, 512, angoW, 256, angoB, tA, 256, nullptr, NB, 256, 512, 1.f, 0);
    gemm(gatt, 512, agoW, 256, nullptr, g1, 256, nullptr, NB3, 256, 512, 1.f, 0);
    add_kernel<<<cdiv((long)NB3 * EE, 256), 256, 0, stream>>>(g_src, g1, (long)NB3 * EE);
    ln_kernel<<<NB, 256, 0, stream>>>(ng_src, tA, n1W, n1B, ng_src);

    // --- second gram on g1 -> g2, ng2 = [g2 | ng_src] ---
    gemm(g1, EE, gp2W, 30, nullptr, gp, 32, nullptr, NB3, 30, EE, 1.f, 0);
    gdirfill_kernel<<<cdiv((long)NB3 * 2, 256), 256, 0, stream>>>(gp, gdir);
    gram_kernel<<<NB, 256, 0, stream>>>(gp, gram, fn);
    gemm(gram, 1024, lg1W, 1024, lg1B, hid, 1024, nullptr, NB, 1024, 1024, 1.f, FLAG_RELU);
    gemm(hid, 1024, lg2W, 256, lg2B, xbuf, 512, nullptr, NB, 256, 1024, 1.f, 0);
    copy_strided<<<cdiv((long)NB * EE, 256), 256, 0, stream>>>(ng_src, EE, xbuf + EE, 512, NB, EE);

    // --- mat3 path: g_src += einsum(g3, mat3) @ l5W ---
    gemm(xbuf, 512, l3W, 1024, l3B, hid, 1024, nullptr, NB, 1024, 512, 1.f, FLAG_RELU);
    gemm(hid, 1024, l4W, 1024, l4B, gram, 1024, fn, NB, 1024, 1024, 1.f, 0);  // mat3 in gram
    gemm(g1, EE, gp3W, 30, nullptr, g3b, 32, nullptr, NB3, 30, EE, 1.f, 0);
    gdirfill_kernel<<<cdiv((long)NB3 * 2, 256), 256, 0, stream>>>(g3b, gdir);
    mat3_apply<<<cdiv((long)NB3 * 32, 256), 256, 0, stream>>>(g3b, gram, gp);
    gemm(gp, 32, l5W, 256, nullptr, g_src, EE, nullptr, NB3, 256, 32, 1.f, FLAG_ADD);

    // --- ng3 path + LN ---
    gemm(xbuf, 512, l1W, 1024, l1B, hid, 1024, nullptr, NB, 1024, 512, 1.f, FLAG_RELU);
    gemm(hid, 1024, l2W, 256, l2B, tA, 256, fn, NB, 256, 1024, 1.f, 0);
    ln_kernel<<<NB, 256, 0, stream>>>(ng_src, tA, n2W, n2B, ng_src);
  }

  // ----- final head -----
  ln_kernel<<<NB, 256, 0, stream>>>(ng_src, nullptr, P(P_fnW), P(P_fnB), ng_src);

  // gram over out_g = [g0 | g_src]  (split-K over ggW rows)
  gemm(g0, GNUM, P(P_ggW), 30, nullptr, gp, 32, nullptr, NB3, 30, GNUM, 1.f, 0);
  gemm(g_src, EE, P(P_ggW) + GNUM * 30, 30, nullptr, gp, 32, nullptr, NB3, 30, EE, 1.f, FLAG_ADD);
  gdirfill_kernel<<<cdiv((long)NB3 * 2, 256), 256, 0, stream>>>(gp, gdir);
  gram_kernel<<<NB, 256, 0, stream>>>(gp, gram, fn);   // fn = fn2

  // out_gg -> attout[:, 0:256)
  gemm(gram, 1024, P(P_g1W), 256, P(P_g1B), hid, 256, nullptr, NB, 256, 1024, 1.f, FLAG_RELU);
  gemm(hid, 256, P(P_g2W), 256, P(P_g2B), attout, 512, nullptr, NB, 256, 256, 1.f, 0);

  // out_ng = relu([ng0|ng_src] @ ng1W + ng1B) @ ng2W + ng2B -> attout[:, 256:512)
  gemm(src + GNUM * 3, FEAT, P(P_ng1W), 256, nullptr, tA, 256, nullptr, NB, 256, NGF, 1.f, 0);
  gemm(ng_src, EE, P(P_ng1W) + NGF * 256, 256, P(P_ng1B), tA, 256, nullptr,
       NB, 256, EE, 1.f, FLAG_ADD | FLAG_RELU);
  gemm(tA, 256, P(P_ng2W), 256, P(P_ng2B), attout + 256, 512, nullptr, NB, 256, 256, 1.f, 0);

  // decode: out = ([out_gg|out_ng] @ decW + decB) / fn2
  decode_kernel<<<NB, 128, 0, stream>>>(attout, P(P_decW), P(P_decB), fn, (float*)d_out);
}
